// Encoder_46548855554754
// MI455X (gfx1250) — matrix operand order repaired and checked by reading, not compiled, not run on hardware
//
#include <hip/hip_runtime.h>
#include <hip/hip_bf16.h>

typedef __bf16 bf16;
typedef __attribute__((ext_vector_type(16))) __bf16 v16bf;
typedef __attribute__((ext_vector_type(8)))  __bf16 v8bf;
typedef __attribute__((ext_vector_type(8)))  float  v8f;

#define LSTM_L 4
#define LSTM_H 512
#define LSTM_F 2048
#define LSTM_B 16
#define LSTM_T 512
#define G4H    (4 * LSTM_H)            // 2048
#define MROWS  (LSTM_B * LSTM_T)       // 8192
#define NWG    32                      // persistent recurrence workgroups

// ---------------------------------------------------------------------------
// fragment helpers (wave32, WMMA 16x16x32 bf16)
// ---------------------------------------------------------------------------
__device__ __forceinline__ v8f vzero8() {
  v8f z;
#pragma unroll
  for (int i = 0; i < 8; ++i) z[i] = 0.0f;
  return z;
}

// A fragment: 16x32 bf16 tile, row-major with leading dim ld (elements).
// Per ISA 7.12.2: row = lane&15 (both halves), lane<16 holds K {0..7,16..23},
// lane>=16 holds K {8..15,24..31}.
__device__ __forceinline__ v16bf load_frag_a(const bf16* p, int ld) {
  const int lane = threadIdx.x & 31;
  const bf16* r = p + (lane & 15) * ld + ((lane >> 4) * 8);
  v8bf lo = *(const v8bf*)(r);
  v8bf hi = *(const v8bf*)(r + 16);
  v16bf a;
#pragma unroll
  for (int j = 0; j < 8; ++j) { a[j] = lo[j]; a[j + 8] = hi[j]; }
  return a;
}

// B fragment from N-major storage: each column's 32 K-values contiguous,
// columns strided by ld elements. col = lane&15, lane<16 holds K {0..7,16..23},
// lane>=16 holds K {8..15,24..31}.
__device__ __forceinline__ v16bf load_frag_b_nmajor(const bf16* p, int ld) {
  const int lane = threadIdx.x & 31;
  const bf16* r = p + (lane & 15) * ld + ((lane >> 4) * 8);
  v8bf lo = *(const v8bf*)(r);
  v8bf hi = *(const v8bf*)(r + 16);
  v16bf b;
#pragma unroll
  for (int j = 0; j < 8; ++j) { b[j] = lo[j]; b[j + 8] = hi[j]; }
  return b;
}

// Grid-wide barrier: monotonic agent-scope counter. Release fence flushes this
// WG's h-slice stores to L2; acquire fence invalidates L0 so next step's h
// loads observe other WGPs' writes.
__device__ __forceinline__ void grid_barrier(unsigned* ctr, unsigned target) {
  __builtin_amdgcn_fence(__ATOMIC_RELEASE, "agent");
  __syncthreads();
  if (threadIdx.x == 0) {
    __hip_atomic_fetch_add(ctr, 1u, __ATOMIC_RELAXED, __HIP_MEMORY_SCOPE_AGENT);
    while (__hip_atomic_load(ctr, __ATOMIC_RELAXED,
                             __HIP_MEMORY_SCOPE_AGENT) < target)
      __builtin_amdgcn_s_sleep(1);
  }
  __syncthreads();
  __builtin_amdgcn_fence(__ATOMIC_ACQUIRE, "agent");
}

// ---------------------------------------------------------------------------
// Generic bf16 WMMA GEMM: C = act(A[M,K] @ B[K,N] + bias[N])
// block tile 64x128, 256 threads = 8 waves (4 M-strips x 2 N-strips of 64)
// ---------------------------------------------------------------------------
#define BM 64
#define BN 128
#define BK 32
#define LDSA 40
#define LDSB 40

template <bool RELU, bool OUTBF>
__global__ __launch_bounds__(256) void gemm_bf16_wmma(
    const bf16* __restrict__ A, const bf16* __restrict__ B,
    const float* __restrict__ bias, void* __restrict__ Cout,
    int N, int K)
{
  __shared__ bf16 sA[BM * LDSA];
  __shared__ bf16 sB[BN * LDSB];   // transposed: sB[n * LDSB + k]

  const int m0 = blockIdx.y * BM;
  const int n0 = blockIdx.x * BN;
  const int tid = threadIdx.x;
  const int wid = tid >> 5;
  const int lane = tid & 31;
  const int wm = (wid & 3) * 16;
  const int wn = (wid >> 2) * 64;

  v8f acc[4];
#pragma unroll
  for (int i = 0; i < 4; ++i) acc[i] = vzero8();

  const int arow = tid >> 2;            // 0..63
  const int acol = (tid & 3) * 8;       // 0,8,16,24
  const int bkrow = tid >> 3;           // 0..31
  const int bnb = (tid & 7) * 16;       // 0..112

  for (int k0 = 0; k0 < K; k0 += BK) {
    // stage A tile (coalesced 16B per thread)
    *(v8bf*)&sA[arow * LDSA + acol] =
        *(const v8bf*)&A[(size_t)(m0 + arow) * K + k0 + acol];
    // stage B tile transposed into N-major LDS
    {
      const bf16* bp = &B[(size_t)(k0 + bkrow) * N + n0 + bnb];
      v8bf b0 = *(const v8bf*)bp;
      v8bf b1 = *(const v8bf*)(bp + 8);
#pragma unroll
      for (int j = 0; j < 8; ++j) {
        sB[(bnb + j) * LDSB + bkrow] = b0[j];
        sB[(bnb + 8 + j) * LDSB + bkrow] = b1[j];
      }
    }
    if (k0 + BK < K) {
      __builtin_prefetch(&A[(size_t)(m0 + arow) * K + k0 + BK + acol], 0, 3);
      __builtin_prefetch(&B[(size_t)(k0 + BK + bkrow) * N + n0 + bnb], 0, 3);
    }
    __syncthreads();

    v16bf af = load_frag_a(&sA[wm * LDSA], LDSA);
#pragma unroll
    for (int nt = 0; nt < 4; ++nt) {
      v16bf bfr = load_frag_b_nmajor(&sB[(wn + nt * 16) * LDSB], LDSB);
      acc[nt] = __builtin_amdgcn_wmma_f32_16x16x32_bf16(
          false, af, false, bfr, (short)0, acc[nt], false, false);
    }
    __syncthreads();
  }

  // epilogue: C layout — VGPR v: lanes 0-15 -> M=v, lanes 16-31 -> M=v+8
#pragma unroll
  for (int nt = 0; nt < 4; ++nt) {
    const int col = n0 + wn + nt * 16 + (lane & 15);
    const float bv = bias ? bias[col] : 0.0f;
#pragma unroll
    for (int v = 0; v < 8; ++v) {
      const int row = m0 + wm + v + ((lane >> 4) * 8);
      float x = acc[nt][v] + bv;
      if (RELU) x = fmaxf(x, 0.0f);
      if (OUTBF) ((bf16*)Cout)[(size_t)row * N + col] = (bf16)x;
      else       ((float*)Cout)[(size_t)row * N + col] = x;
    }
  }
}

// ---------------------------------------------------------------------------
// Persistent per-layer LSTM recurrence.
// grid = 32 WGs (16 h-columns each) x 128 threads (4 waves = i/f/g/o).
// Wh slice lives in LDS for the whole layer; h ping-pongs through L2 with a
// fence-protected grid barrier per step. Requires all 32 WGs co-resident
// (trivially true: MI455X has far more WGPs; 85KB LDS per WG).
// ---------------------------------------------------------------------------
#define LDH 520   // padded leading dim (1040B rows: 16B aligned, bank-spread)

__global__ __launch_bounds__(128) void lstm_layer_persistent(
    const bf16* __restrict__ WhT_l,   // [2048][512] N-major
    const bf16* __restrict__ xg,      // [8192][2048] bf16
    bf16* __restrict__ hall,          // [8192][512] bf16 out
    bf16* __restrict__ hbuf,          // 2 * [16][512] ping-pong
    float* __restrict__ cst,          // [16][512]
    unsigned* __restrict__ ctr,       // barrier counter (pre-zeroed)
    int reverse)
{
  __shared__ bf16 sW[64 * LDH];       // 4 gate tiles x 16 cols, K-contiguous
  __shared__ bf16 sH[16 * LDH];       // staged h_prev
  __shared__ float gates[4][16][16];

  const int tid = threadIdx.x;
  const int w = tid >> 5;             // gate section 0..3
  const int lane = tid & 31;
  const int n0 = blockIdx.x * 16;     // h-column tile
  const int HB = LSTM_B * LSTM_H;

  // preload this WG's Wh slice: col c of gate w -> sW[(w*16+c)*LDH + k]
  for (int i = tid * 8; i < 64 * LSTM_H; i += 128 * 8) {
    const int c = i >> 9;             // 0..63
    const int k = i & (LSTM_H - 1);
    const int gw = c >> 4, cc = c & 15;
    *(v8bf*)&sW[c * LDH + k] =
        *(const v8bf*)&WhT_l[((size_t)(gw * LSTM_H + n0 + cc)) * LSTM_H + k];
  }

  unsigned target = 0;
  for (int k = 0; k < LSTM_T; ++k) {
    const bf16* hp = hbuf + (k & 1) * HB;
    bf16* hn = hbuf + ((k + 1) & 1) * HB;
    const int t = reverse ? (LSTM_T - 1 - k) : k;

    // stage h_prev into LDS (previous step's readers are past the barrier)
    __syncthreads();
    for (int i = tid * 8; i < LSTM_B * LSTM_H; i += 128 * 8) {
      const int m = i >> 9;
      const int kk = i & (LSTM_H - 1);
      *(v8bf*)&sH[m * LDH + kk] = *(const v8bf*)&hp[m * LSTM_H + kk];
    }
    __syncthreads();

    // gate GEMM: two accumulator chains to break WMMA RAW dependence
    v8f acc0 = vzero8(), acc1 = vzero8();
#pragma unroll 2
    for (int kc = 0; kc < LSTM_H; kc += 64) {
      v16bf a0 = load_frag_a(&sH[kc], LDH);
      v16bf b0 = load_frag_b_nmajor(&sW[(w * 16) * LDH + kc], LDH);
      acc0 = __builtin_amdgcn_wmma_f32_16x16x32_bf16(
          false, a0, false, b0, (short)0, acc0, false, false);
      v16bf a1 = load_frag_a(&sH[kc + 32], LDH);
      v16bf b1 = load_frag_b_nmajor(&sW[(w * 16) * LDH + kc + 32], LDH);
      acc1 = __builtin_amdgcn_wmma_f32_16x16x32_bf16(
          false, a1, false, b1, (short)0, acc1, false, false);
    }

    const int gcol0 = w * LSTM_H + n0;
    const int j = lane & 15;
#pragma unroll
    for (int v = 0; v < 8; ++v) {
      const int m = v + ((lane >> 4) * 8);
      gates[w][m][j] = acc0[v] + acc1[v] +
          (float)xg[((size_t)m * LSTM_T + t) * G4H + gcol0 + j];
    }
    __syncthreads();

    for (int e = tid; e < 256; e += 128) {
      const int m = e >> 4, jj = e & 15, col = n0 + jj;
      const float gi = gates[0][m][jj], gf = gates[1][m][jj];
      const float gg = gates[2][m][jj], go = gates[3][m][jj];
      const float cv = cst[m * LSTM_H + col];
      const float si = 1.0f / (1.0f + __expf(-gi));
      const float sf = 1.0f / (1.0f + __expf(-gf));
      const float so = 1.0f / (1.0f + __expf(-go));
      const float cn = sf * cv + si * tanhf(gg);
      const float hn_v = so * tanhf(cn);
      cst[m * LSTM_H + col] = cn;
      const bf16 hb16 = (bf16)hn_v;
      hn[m * LSTM_H + col] = hb16;
      hall[((size_t)m * LSTM_T + t) * LSTM_H + col] = hb16;
    }

    if (k + 1 < LSTM_T) {
      target += gridDim.x;
      grid_barrier(ctr, target);
    }
  }
}

// ---------------------------------------------------------------------------
// casts / transposes / init
// ---------------------------------------------------------------------------
__global__ void cast_f32_bf16_kernel(const float* __restrict__ s,
                                     bf16* __restrict__ d, int n) {
  int i = blockIdx.x * blockDim.x + threadIdx.x;
  const int stride = gridDim.x * blockDim.x;
  for (; i < n; i += stride) d[i] = (bf16)s[i];
}

// Wh [L][H][4H] f32 -> WhT [L][4H][H] bf16 (dst-index iteration: writes coalesce)
__global__ void transpose_wh_kernel(const float* __restrict__ s,
                                    bf16* __restrict__ d) {
  int i = blockIdx.x * blockDim.x + threadIdx.x;
  const int total = LSTM_L * LSTM_H * G4H;
  const int stride = gridDim.x * blockDim.x;
  for (; i < total; i += stride) {
    const int l = i / (LSTM_H * G4H);
    const int r = i % (LSTM_H * G4H);
    const int n = r / LSTM_H;
    const int k = r % LSTM_H;
    d[i] = (bf16)s[(size_t)l * LSTM_H * G4H + (size_t)k * G4H + n];
  }
}

__global__ void zero_state_kernel(bf16* __restrict__ h, float* __restrict__ c,
                                  unsigned* __restrict__ ctr, int n) {
  int i = blockIdx.x * blockDim.x + threadIdx.x;
  const int stride = gridDim.x * blockDim.x;
  for (int j = i; j < 2 * n; j += stride) h[j] = (bf16)0.0f;
  for (int j = i; j < n; j += stride) c[j] = 0.0f;
  if (i == 0) *ctr = 0u;
}

// ---------------------------------------------------------------------------
// host launch
// ---------------------------------------------------------------------------
extern "C" void kernel_launch(void* const* d_in, const int* in_sizes, int n_in,
                              void* d_out, int out_size, void* d_ws,
                              size_t ws_size, hipStream_t stream) {
  (void)in_sizes; (void)n_in; (void)out_size; (void)ws_size;
  const float* x  = (const float*)d_in[0];
  const float* Wx = (const float*)d_in[1];
  const float* Wh = (const float*)d_in[2];
  const float* b  = (const float*)d_in[3];
  const float* W1 = (const float*)d_in[4];
  const float* b1 = (const float*)d_in[5];
  const float* W2 = (const float*)d_in[6];
  const float* b2 = (const float*)d_in[7];

  char* ws = (char*)d_ws;
  size_t off = 0;
  auto alloc = [&](size_t bytes) -> void* {
    void* p = ws + off;
    off = (off + bytes + 255) & ~(size_t)255;
    return p;
  };
  bf16*  Wx_bf = (bf16*)alloc((size_t)LSTM_L * LSTM_H * G4H * sizeof(bf16));
  bf16*  WhT   = (bf16*)alloc((size_t)LSTM_L * G4H * LSTM_H * sizeof(bf16));
  bf16*  W1_bf = (bf16*)alloc((size_t)LSTM_L * LSTM_H * LSTM_F * sizeof(bf16));
  bf16*  W2_bf = (bf16*)alloc((size_t)LSTM_L * LSTM_F * LSTM_H * sizeof(bf16));
  bf16*  xbuf  = (bf16*)alloc((size_t)MROWS * LSTM_H * sizeof(bf16));
  bf16*  xgb   = (bf16*)alloc((size_t)MROWS * G4H * sizeof(bf16));
  bf16*  hall  = (bf16*)alloc((size_t)MROWS * LSTM_H * sizeof(bf16));
  bf16*  mid   = (bf16*)alloc((size_t)MROWS * LSTM_F * sizeof(bf16));
  bf16*  hb    = (bf16*)alloc((size_t)2 * LSTM_B * LSTM_H * sizeof(bf16));
  float* cst   = (float*)alloc((size_t)LSTM_B * LSTM_H * sizeof(float));
  unsigned* bar = (unsigned*)alloc(256);

  const int HB = LSTM_B * LSTM_H;

  cast_f32_bf16_kernel<<<512, 256, 0, stream>>>(x, xbuf, MROWS * LSTM_H);
  cast_f32_bf16_kernel<<<512, 256, 0, stream>>>(Wx, Wx_bf,
                                                LSTM_L * LSTM_H * G4H);
  cast_f32_bf16_kernel<<<512, 256, 0, stream>>>(W1, W1_bf,
                                                LSTM_L * LSTM_H * LSTM_F);
  cast_f32_bf16_kernel<<<512, 256, 0, stream>>>(W2, W2_bf,
                                                LSTM_L * LSTM_F * LSTM_H);
  transpose_wh_kernel<<<512, 256, 0, stream>>>(Wh, WhT);

  const dim3 gridWide(G4H / BN, MROWS / BM);      // N=2048 GEMMs
  const dim3 gridNarrow(LSTM_H / BN, MROWS / BM); // N=512 GEMM

  for (int l = 0; l < LSTM_L; ++l) {
    // xg = x @ Wx[l] + b[l]   -> bf16 [8192, 2048]
    gemm_bf16_wmma<false, true><<<gridWide, 256, 0, stream>>>(
        xbuf, Wx_bf + (size_t)l * LSTM_H * G4H, b + (size_t)l * G4H, xgb,
        G4H, LSTM_H);

    zero_state_kernel<<<32, 256, 0, stream>>>(hb, cst, bar, HB);

    // entire recurrence in one persistent kernel (reverse time for odd layers)
    lstm_layer_persistent<<<NWG, 128, 0, stream>>>(
        WhT + (size_t)l * G4H * LSTM_H, xgb, hall, hb, cst, bar, l & 1);

    // FFN: mid = relu(hall @ W1[l] + b1[l])
    gemm_bf16_wmma<true, true><<<gridWide, 256, 0, stream>>>(
        hall, W1_bf + (size_t)l * LSTM_H * LSTM_F, b1 + (size_t)l * LSTM_F,
        mid, LSTM_F, LSTM_H);

    // out = mid @ W2[l] + b2[l]
    if (l < LSTM_L - 1) {
      gemm_bf16_wmma<false, true><<<gridNarrow, 256, 0, stream>>>(
          mid, W2_bf + (size_t)l * LSTM_F * LSTM_H, b2 + (size_t)l * LSTM_H,
          xbuf, LSTM_H, LSTM_F);
    } else {
      gemm_bf16_wmma<false, false><<<gridNarrow, 256, 0, stream>>>(
          mid, W2_bf + (size_t)l * LSTM_F * LSTM_H, b2 + (size_t)l * LSTM_H,
          d_out, LSTM_H, LSTM_F);
    }
  }
}
